// VQAutoEncoder_71004399337867
// MI455X (gfx1250) — compile-verified, matrix operand order checked by reading, not run on hardware
//
#include <hip/hip_runtime.h>
#include <hip/hip_bf16.h>

typedef __attribute__((ext_vector_type(16))) _Float16 v16h;
typedef __attribute__((ext_vector_type(8)))  float    v8f;

// ---------------------------------------------------------------------------
// Pack two f32 into one dword of two f16 via v_cvt_pk_rtz_f16_f32.
// ---------------------------------------------------------------------------
__device__ __forceinline__ unsigned pack2h(float a, float b) {
  auto p = __builtin_amdgcn_cvt_pkrtz(a, b);           // 2 x f16 in 4 bytes
  return __builtin_bit_cast(unsigned, p);
}

// stage 16 contiguous floats (16B-aligned) into 8 packed-pair LDS dwords
__device__ __forceinline__ void stage_vec16(unsigned* dst, const float* src, float m) {
  const float4* p = (const float4*)src;
  float4 q0 = p[0], q1 = p[1], q2 = p[2], q3 = p[3];
  dst[0] = pack2h(q0.x * m, q0.y * m);
  dst[1] = pack2h(q0.z * m, q0.w * m);
  dst[2] = pack2h(q1.x * m, q1.y * m);
  dst[3] = pack2h(q1.z * m, q1.w * m);
  dst[4] = pack2h(q2.x * m, q2.y * m);
  dst[5] = pack2h(q2.z * m, q2.w * m);
  dst[6] = pack2h(q3.x * m, q3.y * m);
  dst[7] = pack2h(q3.z * m, q3.w * m);
}

// ---------------------------------------------------------------------------
// Gather one wave's 16x32 f16 operand tile from LDS (stored as packed pairs,
// T[row][pair]) into the V_WMMA_F32_16X16X32_F16 per-lane layout
// (CDNA5 ISA): lane L: row = L&15, khalf = L>>4;
//   dword v (v<4)  = pair (v + 4*khalf)        -> K = 8*khalf + 2v, +1
//   dword v (v>=4) = pair (8 + (v-4) + 4*khalf)-> K = 16 + 8*khalf + 2(v-4), +1
// Two 16B LDS reads per operand (ds_load_b128 x2).
// ---------------------------------------------------------------------------
__device__ __forceinline__ v16h gather_pairs(const unsigned (*T)[16], int lane) {
  int m  = lane & 15;
  int kh = lane >> 4;
  uint4 lo = *(const uint4*)&T[m][4 * kh];
  uint4 hi = *(const uint4*)&T[m][8 + 4 * kh];
  struct U { uint4 a, b; } u{lo, hi};
  return __builtin_bit_cast(v16h, u);
}

// ---------------------------------------------------------------------------
// Implicit-GEMM conv2d, templated on kernel size (3x3 or 1x1 here).
// Block = 128 threads = 4 waves. All 4 waves share the weight tile (At);
// each wave computes its own 16(out-ch) x 16(spatial) WMMA tile, so a block
// covers 16 x 64 outputs. Spatial size Ho*Wo is always a multiple of 64 in
// this network, so only Co and K need masking (first conv K=27, last Co=3).
// ---------------------------------------------------------------------------
template <int KHC, int KWC>
__global__ void conv2d_wmma_kernel(const float* __restrict__ in,
                                   const float* __restrict__ wgt,
                                   const float* __restrict__ bias,
                                   const float* __restrict__ resid,
                                   float* __restrict__ out,
                                   int Ci, int Hi, int Wi,
                                   int Co, int Ho, int Wo,
                                   int stride, int padT, int padL) {
  const int tid  = threadIdx.x;
  const int wave = tid >> 5;
  const int lane = tid & 31;
  const int b    = blockIdx.z;
  const int co0  = blockIdx.y * 16;
  const int p0   = blockIdx.x * 64;
  const int HWo  = Ho * Wo;
  const int KK   = KHC * KWC;
  const int Ktot = Ci * KK;
  const bool kAligned = (Ktot & 31) == 0;     // all layers except first conv

  __shared__ unsigned At[16][16];      // weights, packed f16 pairs
  __shared__ unsigned Bt[4][16][16];   // per-wave im2col patches

  const float* inb = in + (size_t)b * Ci * Hi * Wi;

  // this lane stages row (t>>1), k-half (t&1) of its tile
  const int srow = lane >> 1;
  const int shal = lane & 1;

  // spatial position this lane stages for B
  const int pB = p0 + wave * 16 + srow;
  const int oy = pB / Wo, ox = pB - oy * Wo;
  const int iy0 = oy * stride - padT;
  const int ix0 = ox * stride - padL;

  // weight row for staging (clamped for the Co=3 tail)
  const int coA  = co0 + srow;
  const float* wrowbase = wgt + (size_t)min(coA, Co - 1) * Ktot;
  const float  wmask = (coA < Co) ? 1.0f : 0.0f;

  v8f acc = {};

  for (int k0 = 0; k0 < Ktot; k0 += 32) {
    // ---- stage weight tile (wave 0 only) ----
    if (wave == 0) {
      __builtin_prefetch(wrowbase + k0 + 32, 0, 1);    // next chunk -> global_prefetch
      if (kAligned) {
        stage_vec16(&At[srow][shal * 8], wrowbase + k0 + shal * 16, wmask);
      } else {
        float vals[16];
#pragma unroll
        for (int e = 0; e < 16; ++e) {
          int kk = k0 + shal * 16 + e;
          float v = wrowbase[min(kk, Ktot - 1)];
          vals[e] = (kk < Ktot) ? v * wmask : 0.0f;
        }
#pragma unroll
        for (int j = 0; j < 8; ++j)
          At[srow][shal * 8 + j] = pack2h(vals[2 * j], vals[2 * j + 1]);
      }
    }
    // ---- stage im2col tile (every wave its own) ----
    {
      float vals[16];
#pragma unroll
      for (int e = 0; e < 16; ++e) {
        int kk  = k0 + shal * 16 + e;
        int kkc = min(kk, Ktot - 1);
        int ci  = kkc / KK;
        int rem = kkc - ci * KK;
        int ky  = rem / KWC;
        int kx  = rem - ky * KWC;
        int iy  = iy0 + ky;
        int ix  = ix0 + kx;
        int iyc = min(max(iy, 0), Hi - 1);
        int ixc = min(max(ix, 0), Wi - 1);
        float v = inb[(ci * Hi + iyc) * Wi + ixc];
        bool ok = (kk < Ktot) & (iy >= 0) & (iy < Hi) & (ix >= 0) & (ix < Wi);
        vals[e] = ok ? v : 0.0f;
      }
#pragma unroll
      for (int j = 0; j < 8; ++j)
        Bt[wave][srow][shal * 8 + j] = pack2h(vals[2 * j], vals[2 * j + 1]);
    }
    __syncthreads();

    v16h a  = gather_pairs((const unsigned(*)[16])At, lane);
    v16h bv = gather_pairs((const unsigned(*)[16])Bt[wave], lane);
    acc = __builtin_amdgcn_wmma_f32_16x16x32_f16(false, a, false, bv,
                                                 (short)0, acc, false, false);
    __syncthreads();
  }

  // C/D layout: VGPR r, lanes 0-15: M=r, N=lane; lanes 16-31: M=8+r, N=lane-16
  const int n  = lane & 15;
  const int hi = lane >> 4;
  const int p  = p0 + wave * 16 + n;
#pragma unroll
  for (int r = 0; r < 8; ++r) {
    int co = co0 + r + 8 * hi;
    if (co < Co) {
      size_t o = ((size_t)b * Co + co) * HWo + p;
      float v = acc[r] + bias[co];
      if (resid) v += resid[o];
      out[o] = v;
    }
  }
}

// ---------------------------------------------------------------------------
// Generic strided batched GEMM (NT form), no bounds checks:
//   requires M % 16 == 0, N % 64 == 0, K % 32 == 0 (true for all uses here).
//   C[b][m][n] = alpha * sum_k A[b*ab + m*asM + k*asK] * B[b*bb + n*bsN + k*bsK]
// ACONT/BCONT: operand has asK/bsK == 1 (contiguous K) -> float4 loads.
// Block = 128 threads = 4 waves sharing the A tile.
// ---------------------------------------------------------------------------
template <bool ACONT, bool BCONT>
__global__ void gemm_nt_wmma_kernel(const float* __restrict__ A,
                                    const float* __restrict__ B,
                                    float* __restrict__ C,
                                    int K,
                                    long long asM, long long asK, long long ab,
                                    long long bsN, long long bsK, long long bb,
                                    long long csM, long long csN, long long cb,
                                    float alpha) {
  const int tid  = threadIdx.x;
  const int wave = tid >> 5;
  const int lane = tid & 31;
  const int b  = blockIdx.z;
  const int m0 = blockIdx.y * 16;
  const int n0 = blockIdx.x * 64 + wave * 16;

  __shared__ unsigned At[16][16];
  __shared__ unsigned Bt[4][16][16];

  const int srow = lane >> 1;
  const int shal = lane & 1;

  const float* arow = A + (size_t)b * ab + (size_t)(m0 + srow) * asM
                        + (size_t)(shal * 16) * (ACONT ? 1 : asK);
  const float* brow = B + (size_t)b * bb + (size_t)(n0 + srow) * bsN
                        + (size_t)(shal * 16) * (BCONT ? 1 : bsK);
  const long long astep = ACONT ? 32 : 32 * asK;
  const long long bstep = BCONT ? 32 : 32 * bsK;

  v8f acc = {};

  for (int k0 = 0; k0 < K; k0 += 32) {
    if (wave == 0) {
      if (ACONT) {
        stage_vec16(&At[srow][shal * 8], arow, 1.0f);
      } else {
        float vals[16];
#pragma unroll
        for (int e = 0; e < 16; ++e)
          vals[e] = arow[(size_t)e * asK];
#pragma unroll
        for (int j = 0; j < 8; ++j)
          At[srow][shal * 8 + j] = pack2h(vals[2 * j], vals[2 * j + 1]);
      }
    }
    {
      if (BCONT) {
        stage_vec16(&Bt[wave][srow][shal * 8], brow, 1.0f);
      } else {
        float vals[16];
#pragma unroll
        for (int e = 0; e < 16; ++e)
          vals[e] = brow[(size_t)e * bsK];
#pragma unroll
        for (int j = 0; j < 8; ++j)
          Bt[wave][srow][shal * 8 + j] = pack2h(vals[2 * j], vals[2 * j + 1]);
      }
    }
    __syncthreads();

    v16h a  = gather_pairs((const unsigned(*)[16])At, lane);
    v16h bv = gather_pairs((const unsigned(*)[16])Bt[wave], lane);
    acc = __builtin_amdgcn_wmma_f32_16x16x32_f16(false, a, false, bv,
                                                 (short)0, acc, false, false);
    __syncthreads();

    arow += astep;
    brow += bstep;
  }

  const int n  = lane & 15;
  const int hi = lane >> 4;
#pragma unroll
  for (int r = 0; r < 8; ++r) {
    int m = m0 + r + 8 * hi;
    C[(size_t)b * cb + (size_t)m * csM + (size_t)(n0 + n) * csN] = alpha * acc[r];
  }
}

// ---------------------------------------------------------------------------
// GroupNorm (32 groups) with optional fused swish. grid = (B, 32), block 256.
// ---------------------------------------------------------------------------
__global__ void gn_swish_kernel(const float* __restrict__ x,
                                const float* __restrict__ scale,
                                const float* __restrict__ bias,
                                float* __restrict__ out,
                                int C, int HW, int do_swish) {
  const int b = blockIdx.x, g = blockIdx.y;
  const int cpg = C / 32;
  const size_t base = ((size_t)b * C + (size_t)g * cpg) * HW;
  const size_t n = (size_t)cpg * HW;
  const int t = threadIdx.x;

  float sum = 0.0f, sq = 0.0f;
  for (size_t i = t; i < n; i += 256) {
    float v = x[base + i];
    sum += v; sq += v * v;
  }
  __shared__ float r1[256], r2[256];
  r1[t] = sum; r2[t] = sq;
  __syncthreads();
  for (int s = 128; s > 0; s >>= 1) {
    if (t < s) { r1[t] += r1[t + s]; r2[t] += r2[t + s]; }
    __syncthreads();
  }
  float mean = r1[0] / (float)n;
  float var  = r2[0] / (float)n - mean * mean;
  float inv  = rsqrtf(var + 1e-6f);

  for (size_t i = t; i < n; i += 256) {
    int c = g * cpg + (int)(i / HW);
    float v = (x[base + i] - mean) * inv * scale[c] + bias[c];
    if (do_swish) v = v / (1.0f + __expf(-v));
    out[base + i] = v;
  }
}

// row-wise softmax (in place), one block per row of length L
__global__ void softmax_rows_kernel(float* __restrict__ att, int L) {
  float* row = att + (size_t)blockIdx.x * L;
  const int t = threadIdx.x;
  __shared__ float red[256];
  float mx = -3.402823e38f;
  for (int i = t; i < L; i += 256) mx = fmaxf(mx, row[i]);
  red[t] = mx; __syncthreads();
  for (int s = 128; s > 0; s >>= 1) {
    if (t < s) red[t] = fmaxf(red[t], red[t + s]);
    __syncthreads();
  }
  mx = red[0]; __syncthreads();
  float sum = 0.0f;
  for (int i = t; i < L; i += 256) {
    float e = __expf(row[i] - mx);
    row[i] = e; sum += e;
  }
  red[t] = sum; __syncthreads();
  for (int s = 128; s > 0; s >>= 1) {
    if (t < s) red[t] += red[t + s];
    __syncthreads();
  }
  float invs = 1.0f / red[0];
  for (int i = t; i < L; i += 256) row[i] *= invs;
}

// nearest-neighbor 2x upsample, NCHW
__global__ void upsample2x_kernel(const float* __restrict__ in,
                                  float* __restrict__ out,
                                  int C, int H, int W, int total) {
  int i = blockIdx.x * blockDim.x + threadIdx.x;
  if (i >= total) return;
  int W2 = 2 * W, H2 = 2 * H;
  int x = i % W2; int t = i / W2;
  int y = t % H2; t /= H2;
  int c = t % C;  int b = t / C;
  out[i] = in[(((size_t)b * C + c) * H + (y >> 1)) * W + (x >> 1)];
}

// ||codebook[j]||^2
__global__ void rownorm_kernel(const float* __restrict__ cb,
                               float* __restrict__ out, int Kc, int D) {
  int j = blockIdx.x * blockDim.x + threadIdx.x;
  if (j >= Kc) return;
  float s = 0.0f;
  for (int d = 0; d < D; ++d) { float v = cb[(size_t)j * D + d]; s += v * v; }
  out[j] = s;
}

// argmin_j ( ||c_j||^2 - 2 * dot[row][j] ); one block per row
__global__ void argmin_kernel(const float* __restrict__ dot,
                              const float* __restrict__ cbn,
                              int* __restrict__ idx,
                              float* __restrict__ idx_out, int Kc) {
  const float* row = dot + (size_t)blockIdx.x * Kc;
  const int t = threadIdx.x;
  __shared__ float bv[256];
  __shared__ int   bi[256];
  float best = 3.402823e38f; int bj = 0;
  for (int j = t; j < Kc; j += 256) {
    float d = cbn[j] - 2.0f * row[j];
    if (d < best) { best = d; bj = j; }
  }
  bv[t] = best; bi[t] = bj;
  __syncthreads();
  for (int s = 128; s > 0; s >>= 1) {
    if (t < s) {
      if (bv[t + s] < bv[t] || (bv[t + s] == bv[t] && bi[t + s] < bi[t])) {
        bv[t] = bv[t + s]; bi[t] = bi[t + s];
      }
    }
    __syncthreads();
  }
  if (t == 0) { idx[blockIdx.x] = bi[0]; idx_out[blockIdx.x] = (float)bi[0]; }
}

__global__ void zero_kernel(float* p, int n) {
  int i = blockIdx.x * blockDim.x + threadIdx.x;
  if (i < n) p[i] = 0.0f;
}

// straight-through zq + (1+beta)*MSE loss accumulation
__global__ void quant_out_kernel(const float* __restrict__ z,
                                 const float* __restrict__ cbk,
                                 const int* __restrict__ idx,
                                 float* __restrict__ zq,
                                 float* __restrict__ loss,
                                 int D, int HW, int total, float scale) {
  int i = blockIdx.x * blockDim.x + threadIdx.x;
  float d2 = 0.0f;
  if (i < total) {
    int hw = i % HW; int t2 = i / HW;
    int d = t2 % D;  int b  = t2 / D;
    int n = b * HW + hw;
    int j = idx[n];
    float q = cbk[(size_t)j * D + d];
    float f = z[i];
    zq[i] = q;                 // forward value of straight-through estimator
    float df = q - f;
    d2 = df * df * scale;
  }
  __shared__ float red[256];
  red[threadIdx.x] = d2;
  __syncthreads();
  for (int s = 128; s > 0; s >>= 1) {
    if (threadIdx.x < s) red[threadIdx.x] += red[threadIdx.x + s];
    __syncthreads();
  }
  if (threadIdx.x == 0) atomicAdd(loss, red[0]);
}

// ---------------------------------------------------------------------------
// Host orchestration: walk the exact static layer list of build_params().
// ---------------------------------------------------------------------------
static inline int idiv(int a, int b) { return (a + b - 1) / b; }

extern "C" void kernel_launch(void* const* d_in, const int* in_sizes, int n_in,
                              void* d_out, int out_size, void* d_ws, size_t ws_size,
                              hipStream_t stream) {
  (void)in_sizes; (void)out_size; (void)ws_size;
  const int Bn = 2;

  int pi = 0;
  const float* X = (const float*)d_in[pi++];
  auto P = [&]() -> const float* { return (const float*)d_in[pi++]; };
  const float* CBK = (const float*)d_in[n_in - 1];   // codebook is last leaf

  float* fout = (float*)d_out;
  const size_t RECON = (size_t)Bn * 3 * 256 * 256;   // 393216
  float* lossp = fout + RECON;
  float* idxp  = fout + RECON + 1;

  // workspace carve-up (floats)
  float* ws = (float*)d_ws;
  const size_t BUF = 16777216;                       // 2*128*256*256 max activation
  float* pool[4] = { ws, ws + BUF, ws + 2 * BUF, ws + 3 * BUF };
  float* ATT = ws + 4 * BUF;                         // 2*1024*1024
  float* QB  = ATT + 2097152;                        // 2*256*1024 each
  float* KB  = QB + 524288;
  float* VB  = KB + 524288;
  float* HB  = VB + 524288;
  float* DOT = HB + 524288;                          // 2*1024*1024
  float* CBN = DOT + 2097152;                        // 1024
  int*   IDX = (int*)(CBN + 1024);                   // 2048

  int curi = -1;
  const float* cur = X;
  auto scratch = [&](int which) -> float* {
    int c = 0;
    for (int i = 0; i < 4; ++i) {
      if (i == curi) continue;
      if (c == which) return pool[i];
      ++c;
    }
    return pool[0];
  };
  auto setcur = [&](float* p) {
    for (int i = 0; i < 4; ++i) if (pool[i] == p) curi = i;
    cur = p;
  };

  auto conv = [&](const float* in, const float* w, const float* bias,
                  const float* resid, float* out,
                  int Ci, int Hi, int Wi, int Co, int KH,
                  int st, int pT, int pL, int Ho, int Wo) {
    dim3 g((Ho * Wo) / 64, idiv(Co, 16), Bn);
    if (KH == 3)
      conv2d_wmma_kernel<3, 3><<<g, dim3(128), 0, stream>>>(
          in, w, bias, resid, out, Ci, Hi, Wi, Co, Ho, Wo, st, pT, pL);
    else
      conv2d_wmma_kernel<1, 1><<<g, dim3(128), 0, stream>>>(
          in, w, bias, resid, out, Ci, Hi, Wi, Co, Ho, Wo, st, pT, pL);
  };
  auto gn = [&](const float* in, const float* s, const float* bb, float* out,
                int C, int HW, int sw) {
    gn_swish_kernel<<<dim3(Bn, 32), dim3(256), 0, stream>>>(in, s, bb, out, C, HW, sw);
  };
  auto gemm = [&](const float* A, const float* Bm, float* Cm,
                  int M, int N, int Kd,
                  long long asM, long long asK, long long ab,
                  long long bsN, long long bsK, long long bb,
                  long long csM, long long csN, long long cb, float alpha) {
    dim3 g(N / 64, M / 16, Bn);
    if (asK == 1 && bsK == 1)
      gemm_nt_wmma_kernel<true, true><<<g, dim3(128), 0, stream>>>(
          A, Bm, Cm, Kd, asM, asK, ab, bsN, bsK, bb, csM, csN, cb, alpha);
    else if (bsK == 1)
      gemm_nt_wmma_kernel<false, true><<<g, dim3(128), 0, stream>>>(
          A, Bm, Cm, Kd, asM, asK, ab, bsN, bsK, bb, csM, csN, cb, alpha);
    else
      gemm_nt_wmma_kernel<false, false><<<g, dim3(128), 0, stream>>>(
          A, Bm, Cm, Kd, asM, asK, ab, bsN, bsK, bb, csM, csN, cb, alpha);
  };

  auto resblk = [&](int Ci, int Co, int H, int W) {
    const float *n1s = P(), *n1b = P();
    const float *c1w = P(), *c1b = P();
    const float *n2s = P(), *n2b = P();
    const float *c2w = P(), *c2b = P();
    const float *skw = nullptr, *skb = nullptr;
    if (Ci != Co) { skw = P(); skb = P(); }
    float *s1 = scratch(0), *s2 = scratch(1), *s3 = scratch(2);
    int HW = H * W;
    gn(cur, n1s, n1b, s1, Ci, HW, 1);
    conv(s1, c1w, c1b, nullptr, s2, Ci, H, W, Co, 3, 1, 1, 1, H, W);
    gn(s2, n2s, n2b, s1, Co, HW, 1);
    const float* resid = cur;
    if (skw) { conv(cur, skw, skb, nullptr, s3, Ci, H, W, Co, 1, 1, 0, 0, H, W); resid = s3; }
    conv(s1, c2w, c2b, resid, s2, Co, H, W, Co, 3, 1, 1, 1, H, W);
    setcur(s2);
  };

  auto attnblk = [&](int C, int H, int W) {
    const float *ns = P(), *nb = P();
    const float *qw = P(), *qb = P();
    const float *kw = P(), *kb = P();
    const float *vw = P(), *vb = P();
    const float *pw = P(), *pb = P();
    int HW = H * W;
    float *s1 = scratch(0), *s2 = scratch(1);
    gn(cur, ns, nb, s1, C, HW, 0);
    conv(s1, qw, qb, nullptr, QB, C, H, W, C, 1, 1, 0, 0, H, W);
    conv(s1, kw, kb, nullptr, KB, C, H, W, C, 1, 1, 0, 0, H, W);
    conv(s1, vw, vb, nullptr, VB, C, H, W, C, 1, 1, 0, 0, H, W);
    // att[q][k] = scale * sum_c Q[c][q] K[c][k]
    gemm(QB, KB, ATT, HW, HW, C,
         1LL, (long long)HW, (long long)C * HW,
         1LL, (long long)HW, (long long)C * HW,
         (long long)HW, 1LL, (long long)HW * HW,
         rsqrtf((float)C));
    softmax_rows_kernel<<<dim3(Bn * HW), dim3(256), 0, stream>>>(ATT, HW);
    // h[c][q] = sum_k V[c][k] att[q][k]
    gemm(VB, ATT, HB, C, HW, HW,
         (long long)HW, 1LL, (long long)C * HW,
         (long long)HW, 1LL, (long long)HW * HW,
         (long long)HW, 1LL, (long long)C * HW,
         1.0f);
    conv(HB, pw, pb, cur, s2, C, H, W, C, 1, 1, 0, 0, H, W);
    setcur(s2);
  };

  auto down = [&](int C, int H, int W) {   // pad (0,1,0,1), k3 s2 p0
    const float *w = P(), *b = P();
    float* s = scratch(0);
    conv(cur, w, b, nullptr, s, C, H, W, C, 3, 2, 0, 0, H / 2, W / 2);
    setcur(s);
  };

  auto up = [&](int C, int H, int W) {     // nearest x2 then k3 s1 p1
    const float *w = P(), *b = P();
    float *s1 = scratch(0), *s2 = scratch(1);
    int total = Bn * C * 2 * H * 2 * W;
    upsample2x_kernel<<<dim3(idiv(total, 256)), dim3(256), 0, stream>>>(cur, s1, C, H, W, total);
    conv(s1, w, b, nullptr, s2, C, 2 * H, 2 * W, C, 3, 1, 1, 1, 2 * H, 2 * W);
    setcur(s2);
  };

  // ---------------- Encoder ----------------
  int H = 256, W = 256;
  { const float *w = P(), *b = P(); float* s = scratch(0);
    conv(cur, w, b, nullptr, s, 3, H, W, 64, 3, 1, 1, 1, H, W); setcur(s); }
  resblk(64, 64, H, W);  resblk(64, 64, H, W);   down(64, H, W);  H = 128; W = 128;
  resblk(64, 128, H, W); resblk(128, 128, H, W); down(128, H, W); H = 64;  W = 64;
  resblk(128, 128, H, W); resblk(128, 128, H, W); down(128, H, W); H = 32; W = 32;
  resblk(128, 256, H, W); attnblk(256, H, W);
  resblk(256, 256, H, W); attnblk(256, H, W);
  // mid
  resblk(256, 256, H, W); attnblk(256, H, W); resblk(256, 256, H, W);
  { const float *s = P(), *b = P(); float* t = scratch(0);
    gn(cur, s, b, t, 256, H * W, 0); setcur(t); }
  { const float *w = P(), *b = P(); float* t = scratch(0);
    conv(cur, w, b, nullptr, t, 256, H, W, 256, 3, 1, 1, 1, H, W); setcur(t); }

  // ---------------- Quantize ----------------
  const float* Z = cur;          // (B, 256, 32, 32)
  const int D = 256, HWq = 1024, Kc = 1024;
  rownorm_kernel<<<dim3(idiv(Kc, 256)), dim3(256), 0, stream>>>(CBK, CBN, Kc, D);
  // dot[b][hw][j] = sum_d z[b][d][hw] * cbk[j][d]
  gemm(Z, CBK, DOT, HWq, Kc, D,
       1LL, (long long)HWq, (long long)D * HWq,
       (long long)D, 1LL, 0LL,
       (long long)Kc, 1LL, (long long)HWq * Kc,
       1.0f);
  argmin_kernel<<<dim3(Bn * HWq), dim3(256), 0, stream>>>(DOT, CBN, IDX, idxp, Kc);
  zero_kernel<<<dim3(1), dim3(1), 0, stream>>>(lossp, 1);
  {
    float* ZQ = scratch(0);
    int total = Bn * D * HWq;
    float scale = (1.0f + 0.25f) / (float)total;   // (1+beta) * mean
    quant_out_kernel<<<dim3(idiv(total, 256)), dim3(256), 0, stream>>>(
        Z, CBK, IDX, ZQ, lossp, D, HWq, total, scale);
    setcur(ZQ);
  }

  // ---------------- Decoder ----------------
  H = 32; W = 32;
  { const float *w = P(), *b = P(); float* t = scratch(0);
    conv(cur, w, b, nullptr, t, 256, H, W, 256, 3, 1, 1, 1, H, W); setcur(t); }
  resblk(256, 256, H, W); attnblk(256, H, W); resblk(256, 256, H, W);
  // level i=3 (curr_res 32 -> attn after each res)
  resblk(256, 256, H, W); attnblk(256, H, W);
  resblk(256, 256, H, W); attnblk(256, H, W);
  up(256, H, W); H = 64; W = 64;
  // level i=2
  resblk(256, 128, H, W); resblk(128, 128, H, W);
  up(128, H, W); H = 128; W = 128;
  // level i=1
  resblk(128, 128, H, W); resblk(128, 128, H, W);
  up(128, H, W); H = 256; W = 256;
  // level i=0
  resblk(128, 64, H, W); resblk(64, 64, H, W);
  // final norm + conv -> d_out recon region
  { const float *s = P(), *b = P(); float* t = scratch(0);
    gn(cur, s, b, t, 64, H * W, 0); setcur(t); }
  { const float *w = P(), *b = P();
    conv(cur, w, b, nullptr, fout, 64, H, W, 3, 3, 1, 1, 1, H, W); }
}
